// BrainGNN_20091857011532
// MI455X (gfx1250) — matrix operand order repaired and checked by reading, not compiled, not run on hardware
//
#include <hip/hip_runtime.h>
#include <math.h>

// ---------------------------------------------------------------------------
// BrainGNN forward for MI455X (gfx1250, wave32).
// Heavy GEMMs: v_wmma_f32_16x16x32_f16, f32 accumulation.
// B operands pre-swizzled into WMMA fragment order -> b128 vector loads.
// All elementwise strides are powers of two (no integer-division SALU chains).
// ---------------------------------------------------------------------------

typedef __attribute__((ext_vector_type(16))) _Float16 v16h;
typedef __attribute__((ext_vector_type(8)))  _Float16 v8h;
typedef __attribute__((ext_vector_type(8)))  float    v8f;

#define NROI   200
#define KCLUS  8      // cluster dim; 9th "cluster" carries the b2 term (h==1)
#define KCp1   9
#define BGRAPH 512

// ------------------------------- utility -----------------------------------

__global__ void fill_f32(float* p, float v, long n) {
  size_t i = (size_t)blockIdx.x * blockDim.x + threadIdx.x;
  if (i < (size_t)n) p[i] = v;
}

// src f32 [rows, cin] -> dst f16 [rows, 1<<kshift], zero padded
__global__ void cast_pad_f16(const float* src, _Float16* dst, long total,
                             int cin, int kshift) {
  size_t i = (size_t)blockIdx.x * blockDim.x + threadIdx.x;
  if (i >= (size_t)total) return;
  int r = (int)(i >> kshift);
  int c = (int)(i & ((1u << kshift) - 1u));
  dst[i] = (c < cin) ? (_Float16)src[(size_t)r * cin + c] : (_Float16)0.f;
}

// --------------- swizzled-B builders (WMMA fragment order) ------------------
// Block layout: one 512-half block per (ktile, ntile); element t = lane*16+e:
//   K   = ktile*32 + (e>>3)*16 + (lane>>4)*8 + (e&7)
//   col = ntile*16 + (lane&15)
// grid = (ntilesN, nktiles), block = 512.

// Wcat[K, k*cout+o] = w2[(K*cout+o)*8 + k] (k<8) | b2[K*cout+o] (k==8)
__global__ void build_wcat_sw(const float* __restrict__ w2,
                              const float* __restrict__ b2,
                              _Float16* __restrict__ dst,
                              int cin, int cout_shift) {
  const int blk  = blockIdx.y * gridDim.x + blockIdx.x;
  const int t    = threadIdx.x;
  const int lane = t >> 4, e = t & 15;
  const int K    = blockIdx.y * 32 + ((e >> 3) << 4) + ((lane >> 4) << 3) + (e & 7);
  const int col  = blockIdx.x * 16 + (lane & 15);
  const int cout = 1 << cout_shift;
  const int k    = col >> cout_shift;
  const int o    = col & (cout - 1);
  float v = 0.f;
  if (K < cin)
    v = (k < KCLUS) ? w2[(((size_t)K << cout_shift) + o) * KCLUS + k]
                    : b2[((size_t)K << cout_shift) + o];
  dst[(size_t)blk * 512 + t] = (_Float16)v;
}

// B[K, col] = w[col, K]   (w row-major [Ncols, cin]); K >= cin -> 0
__global__ void transpose_sw(const float* __restrict__ w,
                             _Float16* __restrict__ dst, int cin) {
  const int blk  = blockIdx.y * gridDim.x + blockIdx.x;
  const int t    = threadIdx.x;
  const int lane = t >> 4, e = t & 15;
  const int K    = blockIdx.y * 32 + ((e >> 3) << 4) + ((lane >> 4) << 3) + (e & 7);
  const int col  = blockIdx.x * 16 + (lane & 15);
  float v = (K < cin) ? w[(size_t)col * cin + K] : 0.f;
  dst[(size_t)blk * 512 + t] = (_Float16)v;
}

// H[n,k] = relu(sum_i pos[n,i]*w1[k,i]) for k<8 ; H[n,8]=1. Thread per (n, k16).
__global__ void compute_h(const float* __restrict__ pos,
                          const float* __restrict__ w1,
                          float* __restrict__ H, int Mn) {
  size_t i = (size_t)blockIdx.x * blockDim.x + threadIdx.x;
  if (i >= ((size_t)Mn << 4)) return;
  int n = (int)(i >> 4), k = (int)(i & 15);
  if (k >= KCp1) return;
  if (k == KCLUS) { H[(size_t)n * KCp1 + k] = 1.f; return; }
  const float* pr = pos + (size_t)n * NROI;
  const float* wr = w1 + (size_t)k * NROI;
  float acc = 0.f;
  for (int c = 0; c < NROI; ++c) acc += pr[c] * wr[c];
  H[(size_t)n * KCp1 + k] = acc > 0.f ? acc : 0.f;
}

// -------------- WMMA GEMM with fused cluster contraction --------------------
// One wave per 16-row x 16-col(cout) tile. Keeps 9 accumulators (one per
// cluster); A fragment loaded once per k-step and reused for 9 WMMAs; epilogue
// contracts with h[n,k] in registers and stores xt directly (no atomics).
__global__ void wmma_xt_kernel(const _Float16* __restrict__ A,
                               const _Float16* __restrict__ Bsw,
                               const float* __restrict__ H,
                               float* __restrict__ xt,
                               int Kpad, int cout_shift) {
  const int tileC = blockIdx.x;            // gridDim.x = cout/16
  const int ntC   = gridDim.x;
  const int ntTot = KCp1 * ntC;            // total column tiles in Bsw
  const int tileM = blockIdx.y;
  const int lane  = threadIdx.x & 31;
  const int row   = lane & 15, half = lane >> 4;
  const _Float16* Arow = A + (size_t)(tileM * 16 + row) * Kpad;

  v8f zero = {};
  v8f acc[KCp1];
#pragma unroll
  for (int k = 0; k < KCp1; ++k) acc[k] = zero;

  for (int kt = 0; kt < Kpad; kt += 32) {
    if (kt + 32 < Kpad) __builtin_prefetch(Arow + kt + 32, 0, 1);  // global_prefetch
    const int aoff = kt + half * 8;
    v8h alo = *(const v8h*)(Arow + aoff);
    v8h ahi = *(const v8h*)(Arow + aoff + 16);
    v16h a = __builtin_shufflevector(alo, ahi, 0, 1, 2, 3, 4, 5, 6, 7,
                                     8, 9, 10, 11, 12, 13, 14, 15);
    const _Float16* bbase =
        Bsw + (size_t)(kt >> 5) * ntTot * 512 + (size_t)lane * 16;
#pragma unroll
    for (int k = 0; k < KCp1; ++k) {
      const _Float16* bp = bbase + (size_t)(k * ntC + tileC) * 512;
      v8h blo = *(const v8h*)bp;
      v8h bhi = *(const v8h*)(bp + 8);
      v16h b = __builtin_shufflevector(blo, bhi, 0, 1, 2, 3, 4, 5, 6, 7,
                                       8, 9, 10, 11, 12, 13, 14, 15);
      acc[k] = __builtin_amdgcn_wmma_f32_16x16x32_f16(false, a, false, b,
                                                      (short)0, acc[k],
                                                      false, false);
    }
  }
#pragma unroll
  for (int r = 0; r < 8; ++r) {
    int n = tileM * 16 + half * 8 + r;
    const float* hrow = H + (size_t)n * KCp1;
    float o = 0.f;
#pragma unroll
    for (int k = 0; k < KCp1; ++k) o += hrow[k] * acc[k][r];
    xt[((size_t)n << cout_shift) + tileC * 16 + row] = o;
  }
}

// Plain WMMA GEMM + bias: C[M,Ncols] = A[M,Kpad]*B[Kpad,Ncols] + bias
__global__ void wmma_gemm_bias(const _Float16* __restrict__ A,
                               const _Float16* __restrict__ Bsw,
                               const float* __restrict__ bias,
                               float* __restrict__ C, int Kpad, int Ncols) {
  const int tileN = blockIdx.x, tileM = blockIdx.y;
  const int ntN = gridDim.x;
  const int lane = threadIdx.x & 31;
  const int row  = lane & 15, half = lane >> 4;
  const _Float16* Arow = A + (size_t)(tileM * 16 + row) * Kpad;
  const int coln = tileN * 16 + row;
  v8f acc = {};
  for (int kt = 0; kt < Kpad; kt += 32) {
    const int aoff = kt + half * 8;
    v8h alo = *(const v8h*)(Arow + aoff);
    v8h ahi = *(const v8h*)(Arow + aoff + 16);
    v16h a = __builtin_shufflevector(alo, ahi, 0, 1, 2, 3, 4, 5, 6, 7,
                                     8, 9, 10, 11, 12, 13, 14, 15);
    const _Float16* bp =
        Bsw + ((size_t)(kt >> 5) * ntN + tileN) * 512 + (size_t)lane * 16;
    v8h blo = *(const v8h*)bp;
    v8h bhi = *(const v8h*)(bp + 8);
    v16h b = __builtin_shufflevector(blo, bhi, 0, 1, 2, 3, 4, 5, 6, 7,
                                     8, 9, 10, 11, 12, 13, 14, 15);
    acc = __builtin_amdgcn_wmma_f32_16x16x32_f16(false, a, false, b,
                                                 (short)0, acc, false, false);
  }
#pragma unroll
  for (int r = 0; r < 8; ++r) {
    int m = half * 8 + r;
    C[(size_t)(tileM * 16 + m) * Ncols + coln] = acc[r] + bias[coln];
  }
}

// ------------------------------ edge kernels --------------------------------

__global__ void edge_agg(const int* __restrict__ ei, const float* __restrict__ ew,
                         const float* __restrict__ em, const float* __restrict__ xt,
                         float* __restrict__ agg, float* __restrict__ cnt,
                         int E, int dshift) {
  size_t i = (size_t)blockIdx.x * blockDim.x + threadIdx.x;
  if (i >= ((size_t)E << dshift)) return;
  int e = (int)(i >> dshift);
  int c = (int)(i & ((1u << dshift) - 1u));
  float m = em[e];
  if (m == 0.f) return;
  int s = ei[e], dn = ei[(size_t)E + e];
  float w = ew[e] * m;
  atomicAdd(&agg[((size_t)dn << dshift) + c], w * xt[((size_t)s << dshift) + c]);
  if (c == 0) atomicAdd(&cnt[dn], m);
}

__global__ void mean_bias(const float* agg, const float* cnt, const float* bias,
                          float* out, long total, int dshift) {
  size_t i = (size_t)blockIdx.x * blockDim.x + threadIdx.x;
  if (i >= (size_t)total) return;
  int n = (int)(i >> dshift);
  int c = (int)(i & ((1u << dshift) - 1u));
  out[i] = agg[i] / fmaxf(cnt[n], 1.f) + bias[c];
}

__global__ void edge_remap(const int* ei, const float* em, const float* sel,
                           const int* newid, int* ei_new, float* em_new, int E) {
  size_t e = (size_t)blockIdx.x * blockDim.x + threadIdx.x;
  if (e >= (size_t)E) return;
  int s = ei[e], d = ei[(size_t)E + e];
  em_new[e] = em[e] * sel[s] * sel[d];
  ei_new[e] = newid[s];
  ei_new[(size_t)E + e] = newid[d];
}

// ------------------------------- top-k pool ---------------------------------
__global__ void topk_pool_kernel(const float* __restrict__ x,
                                 const float* __restrict__ pos_prev,
                                 const float* __restrict__ wpool,
                                 int n_per, int kkeep, int d,
                                 float* x_new, float* pos_new,
                                 float* sel, int* newid,
                                 float* vals_out, float* sraw_out) {
  const int g = blockIdx.x, t = threadIdx.x;
  __shared__ float ss[256];
  __shared__ int   si[256];
  __shared__ float red[256];
  __shared__ float wnorm;
  red[t] = (t < d) ? wpool[t] * wpool[t] : 0.f;
  __syncthreads();
  for (int st = 128; st > 0; st >>= 1) {
    if (t < st) red[t] += red[t + st];
    __syncthreads();
  }
  if (t == 0) wnorm = sqrtf(red[0]);
  __syncthreads();

  if (t < n_per) {
    const float* xr = x + (size_t)(g * n_per + t) * d;
    float acc = 0.f;
    for (int c = 0; c < d; ++c) acc += xr[c] * wpool[c];
    ss[t] = acc / wnorm;
    si[t] = t;
    sel[(size_t)g * n_per + t]   = 0.f;
    newid[(size_t)g * n_per + t] = 0;
  } else {
    ss[t] = -INFINITY;
    si[t] = 0x7fffffff;
  }

  // bitonic sort: descending score, ascending index on ties (jax.lax.top_k)
  for (int ksz = 2; ksz <= 256; ksz <<= 1) {
    for (int j = ksz >> 1; j > 0; j >>= 1) {
      __syncthreads();
      int p = t ^ j;
      if (p > t) {
        bool desc = ((t & ksz) == 0);
        float as = ss[t], bs = ss[p];
        int   ai = si[t], bi = si[p];
        bool aFirst = (as > bs) || (as == bs && ai < bi);
        if (desc ? !aFirst : aFirst) {
          ss[t] = bs; ss[p] = as;
          si[t] = bi; si[p] = ai;
        }
      }
    }
  }
  __syncthreads();

  if (t < kkeep) {
    float raw = ss[t];
    int local = si[t];
    int node  = g * n_per + local;
    float sg  = 1.f / (1.f + expf(-raw));
    int nn    = g * kkeep + t;
    sel[node]   = 1.f;
    newid[node] = nn;
    vals_out[nn] = sg;
    if (sraw_out) sraw_out[nn] = raw;
    const float* xr = x + (size_t)node * d;
    float* xo = x_new + (size_t)nn * d;
    for (int c = 0; c < d; ++c) xo[c] = xr[c] * sg;
    const float* pr = pos_prev + (size_t)node * NROI;
    float* po = pos_new + (size_t)nn * NROI;
    for (int c = 0; c < NROI; ++c) po[c] = pr[c];
  }
}

// ----------------------------- readout / MLP --------------------------------

__global__ void readout_kernel(const float* X, float* R, int kkeep, int d,
                               int coff, int ldr) {
  int g = blockIdx.x, c = threadIdx.x;
  if (c >= d) return;
  float mx = -INFINITY, sm = 0.f;
  for (int j = 0; j < kkeep; ++j) {
    float v = X[((size_t)g * kkeep + j) * d + c];
    mx = fmaxf(mx, v);
    sm += v;
  }
  R[(size_t)g * ldr + coff + c]     = mx;
  R[(size_t)g * ldr + coff + d + c] = sm / (float)kkeep;
}

// PReLU + BatchNorm1d (train stats, biased var). One block per column.
__global__ void prelu_bn_kernel(const float* Hm, const float* alpha_p,
                                const float* gamma, const float* beta,
                                float* Z, int cols) {
  const int c = blockIdx.x, t = threadIdx.x;  // rows == 512 == 2*256
  __shared__ float r1[256], r2[256];
  float alpha = alpha_p[0];
  float v[2], s = 0.f, s2 = 0.f;
#pragma unroll
  for (int i = 0; i < 2; ++i) {
    int rIdx = t + i * 256;
    float hv = Hm[(size_t)rIdx * cols + c];
    hv = hv > 0.f ? hv : alpha * hv;
    v[i] = hv; s += hv; s2 += hv * hv;
  }
  r1[t] = s; r2[t] = s2;
  __syncthreads();
  for (int st = 128; st > 0; st >>= 1) {
    if (t < st) { r1[t] += r1[t + st]; r2[t] += r2[t + st]; }
    __syncthreads();
  }
  float mu  = r1[0] / 512.f;
  float var = r2[0] / 512.f - mu * mu;
  float inv = rsqrtf(var + 1e-5f);
#pragma unroll
  for (int i = 0; i < 2; ++i) {
    int rIdx = t + i * 256;
    Z[(size_t)rIdx * cols + c] = (v[i] - mu) * inv * gamma[c] + beta[c];
  }
}

__global__ void mlp3_logsoftmax(const float* Z, const float* w3, const float* b3,
                                float* out, int d) {
  int g = blockIdx.x * blockDim.x + threadIdx.x;
  if (g >= BGRAPH) return;
  float l0 = b3[0], l1 = b3[1];
  const float* zr = Z + (size_t)g * d;
  for (int c = 0; c < d; ++c) {
    l0 += zr[c] * w3[c];
    l1 += zr[c] * w3[d + c];
  }
  float m = fmaxf(l0, l1);
  float lse = m + logf(expf(l0 - m) + expf(l1 - m));
  out[g * 2 + 0] = l0 - lse;
  out[g * 2 + 1] = l1 - lse;
}

// ------------------------------- launcher -----------------------------------

static inline unsigned cdiv(long a, long b) { return (unsigned)((a + b - 1) / b); }
static inline int shift_pow2(int x) { int s = 0; while ((1 << s) < x) ++s; return s; }

extern "C" void kernel_launch(void* const* d_in, const int* in_sizes, int n_in,
                              void* d_out, int out_size, void* d_ws, size_t ws_size,
                              hipStream_t stream) {
  (void)n_in; (void)out_size; (void)ws_size;
  // ---- inputs (setup_inputs order; params leaves in sorted pytree order) ----
  const float* x0   = (const float*)d_in[0];
  const float* ea   = (const float*)d_in[1];
  const float* pos0 = (const float*)d_in[2];
  const int*   ei0  = (const int*)d_in[3];
  const float* c1_b2 = (const float*)d_in[5],  *c1_bias = (const float*)d_in[6];
  const float* c1_w1 = (const float*)d_in[7],  *c1_w2   = (const float*)d_in[8];
  const float* c2_b2 = (const float*)d_in[9],  *c2_bias = (const float*)d_in[10];
  const float* c2_w1 = (const float*)d_in[11], *c2_w2   = (const float*)d_in[12];
  const float* c3_b2 = (const float*)d_in[13], *c3_bias = (const float*)d_in[14];
  const float* c3_w1 = (const float*)d_in[15], *c3_w2   = (const float*)d_in[16];
  const float* m1_b  = (const float*)d_in[17], *m1_beta = (const float*)d_in[18];
  const float* m1_g  = (const float*)d_in[19], *m1_pr   = (const float*)d_in[20];
  const float* m1_w  = (const float*)d_in[21];
  const float* m2_b  = (const float*)d_in[22], *m2_beta = (const float*)d_in[23];
  const float* m2_g  = (const float*)d_in[24], *m2_pr   = (const float*)d_in[25];
  const float* m2_w  = (const float*)d_in[26];
  const float* m3_b  = (const float*)d_in[27], *m3_w    = (const float*)d_in[28];
  const float* pool1 = (const float*)d_in[29];
  const float* pool2 = (const float*)d_in[30];
  const float* pool3 = (const float*)d_in[31];

  const int E  = in_sizes[1];
  const int M0 = in_sizes[0] / NROI;      // B*NROI*DIM_IN / NROI ... == 102400
  const int M1 = M0 / 2, M2 = M0 / 4, M3 = M0 / 8;
  const int D1c = 64, D2c = 64, D3c = 128;

  float* out = (float*)d_out;
  float* o_s1  = out + 1024;
  float* o_s2  = o_s1 + (size_t)BGRAPH * 100;
  float* o_s3  = o_s2 + (size_t)BGRAPH * 50;
  float* o_s1r = o_s3 + (size_t)BGRAPH * 25;

  // ---- workspace bump allocator (deterministic layout every call) ----
  char* wp = (char*)d_ws;
  auto alloc = [&](size_t bytes) -> void* {
    void* p = (void*)wp;
    wp += (bytes + 255) & ~(size_t)255;
    return p;
  };
  _Float16* AH = (_Float16*)alloc((size_t)M0 * 256 * 2);     // A operand (reused)
  _Float16* WH = (_Float16*)alloc((size_t)8 * 72 * 512 * 2); // swizzled B (reused)
  float* HF  = (float*)alloc((size_t)M0 * KCp1 * 4);
  float* XT  = (float*)alloc((size_t)M0 * 64 * 4);
  float* AGG = (float*)alloc((size_t)M0 * 64 * 4);
  float* CNT = (float*)alloc((size_t)M0 * 4);
  float* HL  = (float*)alloc((size_t)M0 * 64 * 4);
  float* XP1 = (float*)alloc((size_t)M1 * D1c * 4);
  float* XP2 = (float*)alloc((size_t)M2 * D2c * 4);
  float* XP3 = (float*)alloc((size_t)M3 * D3c * 4);
  float* POS1 = (float*)alloc((size_t)M1 * NROI * 4);
  float* POS2 = (float*)alloc((size_t)M2 * NROI * 4);
  float* POS3 = (float*)alloc((size_t)M3 * NROI * 4);
  float* SEL  = (float*)alloc((size_t)M0 * 4);
  int*   NEW  = (int*)alloc((size_t)M0 * 4);
  float* EM0  = (float*)alloc((size_t)E * 4);
  int*   EI1  = (int*)alloc((size_t)2 * E * 4);
  float* EM1  = (float*)alloc((size_t)E * 4);
  int*   EI2  = (int*)alloc((size_t)2 * E * 4);
  float* EM2  = (float*)alloc((size_t)E * 4);
  float* Rm   = (float*)alloc((size_t)BGRAPH * 512 * 4);
  float* GM   = (float*)alloc((size_t)BGRAPH * 256 * 4);
  float* Z1   = (float*)alloc((size_t)BGRAPH * 128 * 4);
  float* Z2   = (float*)alloc((size_t)BGRAPH * 256 * 4);

  const int BS = 256;
  fill_f32<<<cdiv((long)E, BS), BS, 0, stream>>>(EM0, 1.f, E);

  // ---------- one Ra-GNN conv layer ----------
  auto gconv = [&](const float* xin, const float* posin, const int* ei, const float* em,
                   const float* w1, const float* w2, const float* b2, const float* bias,
                   int Mn, int cin, int cout, float* hout) {
    const int kshift = (cin == 200) ? 8 : shift_pow2(cin);  // pad 200 -> 256
    const int kpad   = 1 << kshift;
    const int cshift = shift_pow2(cout);
    cast_pad_f16<<<cdiv((long)Mn << kshift, BS), BS, 0, stream>>>(
        xin, AH, (long)Mn << kshift, cin, kshift);
    {
      dim3 g(KCp1 * cout / 16, kpad / 32);
      build_wcat_sw<<<g, 512, 0, stream>>>(w2, b2, WH, cin, cshift);
    }
    compute_h<<<cdiv((long)Mn << 4, BS), BS, 0, stream>>>(posin, w1, HF, Mn);
    {
      dim3 g(cout / 16, Mn / 16);
      wmma_xt_kernel<<<g, 32, 0, stream>>>(AH, WH, HF, XT, kpad, cshift);
    }
    fill_f32<<<cdiv((long)Mn << cshift, BS), BS, 0, stream>>>(AGG, 0.f, (long)Mn << cshift);
    fill_f32<<<cdiv((long)Mn, BS), BS, 0, stream>>>(CNT, 0.f, Mn);
    edge_agg<<<cdiv((long)E << cshift, BS), BS, 0, stream>>>(ei, ea, em, XT, AGG, CNT, E, cshift);
    mean_bias<<<cdiv((long)Mn << cshift, BS), BS, 0, stream>>>(
        AGG, CNT, bias, hout, (long)Mn << cshift, cshift);
  };

  // ---------- layer 1 ----------
  gconv(x0, pos0, ei0, EM0, c1_w1, c1_w2, c1_b2, c1_bias, M0, NROI, D1c, HL);
  topk_pool_kernel<<<BGRAPH, 256, 0, stream>>>(HL, pos0, pool1, NROI, 100, D1c,
                                               XP1, POS1, SEL, NEW, o_s1, o_s1r);
  edge_remap<<<cdiv((long)E, BS), BS, 0, stream>>>(ei0, EM0, SEL, NEW, EI1, EM1, E);

  // ---------- layer 2 ----------
  gconv(XP1, POS1, EI1, EM1, c2_w1, c2_w2, c2_b2, c2_bias, M1, D1c, D2c, HL);
  topk_pool_kernel<<<BGRAPH, 256, 0, stream>>>(HL, POS1, pool2, 100, 50, D2c,
                                               XP2, POS2, SEL, NEW, o_s2, nullptr);
  edge_remap<<<cdiv((long)E, BS), BS, 0, stream>>>(EI1, EM1, SEL, NEW, EI2, EM2, E);

  // ---------- layer 3 ----------
  gconv(XP2, POS2, EI2, EM2, c3_w1, c3_w2, c3_b2, c3_bias, M2, D2c, D3c, HL);
  topk_pool_kernel<<<BGRAPH, 256, 0, stream>>>(HL, POS2, pool3, 50, 25, D3c,
                                               XP3, POS3, SEL, NEW, o_s3, nullptr);
  // ei3/em3 are dead in the reference outputs -> skip remap

  // ---------- readout ----------
  readout_kernel<<<BGRAPH, 64, 0, stream>>>(XP1, Rm, 100, D1c, 0, 512);
  readout_kernel<<<BGRAPH, 64, 0, stream>>>(XP2, Rm, 50, D2c, 128, 512);
  readout_kernel<<<BGRAPH, 128, 0, stream>>>(XP3, Rm, 25, D3c, 256, 512);

  // ---------- MLP head ----------
  // mlp1: [512,512] x [512,128]
  cast_pad_f16<<<cdiv((long)BGRAPH << 9, BS), BS, 0, stream>>>(
      Rm, AH, (long)BGRAPH << 9, 512, 9);
  {
    dim3 g(128 / 16, 512 / 32);
    transpose_sw<<<g, 512, 0, stream>>>(m1_w, WH, 512);
  }
  {
    dim3 g(128 / 16, BGRAPH / 16);
    wmma_gemm_bias<<<g, 32, 0, stream>>>(AH, WH, m1_b, GM, 512, 128);
  }
  prelu_bn_kernel<<<128, 256, 0, stream>>>(GM, m1_pr, m1_g, m1_beta, Z1, 128);

  // mlp2: [512,128] x [128,256]
  cast_pad_f16<<<cdiv((long)BGRAPH << 7, BS), BS, 0, stream>>>(
      Z1, AH, (long)BGRAPH << 7, 128, 7);
  {
    dim3 g(256 / 16, 128 / 32);
    transpose_sw<<<g, 512, 0, stream>>>(m2_w, WH, 128);
  }
  {
    dim3 g(256 / 16, BGRAPH / 16);
    wmma_gemm_bias<<<g, 32, 0, stream>>>(AH, WH, m2_b, GM, 128, 256);
  }
  prelu_bn_kernel<<<256, 256, 0, stream>>>(GM, m2_pr, m2_g, m2_beta, Z2, 256);

  // mlp3 + log_softmax
  mlp3_logsoftmax<<<cdiv(BGRAPH, BS), BS, 0, stream>>>(Z2, m3_w, m3_b, out, 256);
}
